// SelfAttention_65738769432686
// MI455X (gfx1250) — matrix operand order repaired and checked by reading, not compiled, not run on hardware
//
#include <hip/hip_runtime.h>
#include <hip/hip_bf16.h>

// ---------------- problem constants ----------------
#define BB    2
#define SS    2048
#define DD    2048
#define HQN   16
#define HKVN  4
#define HDIM  128
#define MROWS (BB * SS)          // 4096

typedef __attribute__((ext_vector_type(16))) __bf16       bf16x16;
typedef __attribute__((ext_vector_type(8)))  float        f32x8;
typedef __attribute__((ext_vector_type(4)))  unsigned int u32x4;

union Frag16 { bf16x16 v; u32x4 u[2]; };

__device__ __forceinline__ f32x8 wmma_bf16(bf16x16 a, bf16x16 b, f32x8 c) {
  return __builtin_amdgcn_wmma_f32_16x16x32_bf16(false, a, false, b, (short)0, c, false, false);
}

__device__ __forceinline__ void wait_async0() {
  asm volatile("s_wait_asynccnt 0x0" ::: "memory");
}

// ---------------- LayerNorm (fp32 math, bf16 in/out) ----------------
__global__ __launch_bounds__(256, 1)
void ln_kernel(const __hip_bfloat16* __restrict__ x, const float* __restrict__ g,
               __hip_bfloat16* __restrict__ xn) {
  const int row = blockIdx.x;
  const __hip_bfloat16* xr = x + (size_t)row * DD;
  __hip_bfloat16* outr = xn + (size_t)row * DD;
  float s = 0.f, s2 = 0.f;
  for (int i = threadIdx.x; i < DD; i += 256) {
    float v = __bfloat162float(xr[i]);
    s += v; s2 += v * v;
  }
  __shared__ float red0[256], red1[256];
  red0[threadIdx.x] = s; red1[threadIdx.x] = s2;
  __syncthreads();
  for (int off = 128; off > 0; off >>= 1) {
    if ((int)threadIdx.x < off) {
      red0[threadIdx.x] += red0[threadIdx.x + off];
      red1[threadIdx.x] += red1[threadIdx.x + off];
    }
    __syncthreads();
  }
  const float mu = red0[0] * (1.0f / DD);
  const float var = red1[0] * (1.0f / DD) - mu * mu;
  const float rstd = rsqrtf(var + 1e-5f);
  for (int i = threadIdx.x; i < DD; i += 256) {
    float v = (__bfloat162float(xr[i]) - mu) * rstd * g[i];
    outr[i] = __float2bfloat16(v);
  }
}

// ---------------- bf16 GEMM: C[M,N] = A[M,K] @ W[K,N] ----------------
// Block tile 256x128, BK=32, double-buffered LDS with async-to-LDS A staging.
// 8 waves: 4 along M x 2 along N, each wave computes 64x64 (16 WMMA tiles).
#define LDT 40   // 32 + 8 pad (80B row stride keeps 16B alignment)

__global__ __launch_bounds__(256, 1)
void gemm_bf16(const __hip_bfloat16* __restrict__ A, const __hip_bfloat16* __restrict__ W,
               __hip_bfloat16* __restrict__ C, int M, int N, int K) {
  __shared__ __hip_bfloat16 sA[2][256][LDT];   // [m][k]
  __shared__ __hip_bfloat16 sB[2][128][LDT];   // [n][k]  (W tile transposed)
  const int t = threadIdx.x;
  const int wid = t >> 5, lane = t & 31;
  const int half = lane >> 4, ln = lane & 15;
  const int m0 = blockIdx.x * 256;
  const int n0 = blockIdx.y * 128;
  const int wm = (wid & 3) * 64;
  const int wn = (wid >> 2) * 64;

  f32x8 acc[4][4] = {};
  const int ntiles = K >> 5;

  auto stage = [&](int it, int nb) {
    const int k0 = it << 5;
    // A tile 256x32: one row per thread, 64B via async DMA to LDS.
    {
      const __hip_bfloat16* sbase = A + (size_t)m0 * K + k0;
      const unsigned voff = (unsigned)(t * K * 2);               // byte offset per lane
      const unsigned ldsb = (unsigned)(size_t)&sA[nb][t][0];     // LDS byte address
      asm volatile(
        "global_load_async_to_lds_b128 %0, %1, %2 offset:0\n\t"
        "global_load_async_to_lds_b128 %0, %1, %2 offset:16\n\t"
        "global_load_async_to_lds_b128 %0, %1, %2 offset:32\n\t"
        "global_load_async_to_lds_b128 %0, %1, %2 offset:48"
        :: "v"(ldsb), "v"(voff), "s"(sbase) : "memory");
    }
    // W tile 32x128 transposed into [n][k]: manual scatter (DMA cannot transpose).
    {
      const int kk = t >> 3, c0 = (t & 7) * 16;
      const __hip_bfloat16* gp = W + (size_t)(k0 + kk) * N + n0 + c0;
      __builtin_prefetch(gp + 32 * (size_t)N, 0, 1);   // next K-tile -> global_prefetch_b8
      u32x4 u0 = *(const u32x4*)gp;
      u32x4 u1 = *(const u32x4*)(gp + 8);
      alignas(16) __hip_bfloat16 tmp[16];
      *(u32x4*)&tmp[0] = u0; *(u32x4*)&tmp[8] = u1;
#pragma unroll
      for (int i = 0; i < 16; i++) sB[nb][c0 + i][kk] = tmp[i];
    }
  };

  stage(0, 0);
  for (int it = 0; it < ntiles; ++it) {
    const int cur = it & 1;
    wait_async0();          // this wave's DMA for buf[cur] done
    __syncthreads();        // everyone's staging for buf[cur] visible
    if (it + 1 < ntiles) stage(it + 1, cur ^ 1);   // overlap next DMA with WMMAs

    Frag16 af[4], bfr[4];
#pragma unroll
    for (int i = 0; i < 4; i++) {
      const __hip_bfloat16* p = &sA[cur][wm + i * 16 + ln][half * 8];
      af[i].u[0] = *(const u32x4*)p;
      af[i].u[1] = *(const u32x4*)(p + 16);
    }
#pragma unroll
    for (int j = 0; j < 4; j++) {
      const __hip_bfloat16* p = &sB[cur][wn + j * 16 + ln][half * 8];
      bfr[j].u[0] = *(const u32x4*)p;
      bfr[j].u[1] = *(const u32x4*)(p + 16);
    }
#pragma unroll
    for (int i = 0; i < 4; i++)
#pragma unroll
      for (int j = 0; j < 4; j++)
        acc[i][j] = wmma_bf16(af[i].v, bfr[j].v, acc[i][j]);
  }

#pragma unroll
  for (int i = 0; i < 4; i++)
#pragma unroll
    for (int j = 0; j < 4; j++) {
      const int row0 = m0 + wm + i * 16 + half * 8;
      const int col  = n0 + wn + j * 16 + ln;
#pragma unroll
      for (int r = 0; r < 8; r++)
        C[(size_t)(row0 + r) * N + col] = __float2bfloat16(acc[i][j][r]);
    }
}

// ---------------- RoPE (interleaved pairs, fp32) ----------------
__global__ __launch_bounds__(256, 1)
void rope_kernel(__hip_bfloat16* __restrict__ q, int H, int total_pairs) {
  const int idx = blockIdx.x * 256 + threadIdx.x;
  if (idx >= total_pairs) return;
  const int i = idx & 63;                  // HD/2 = 64 pairs
  const int s = (idx >> 6) / H % SS;       // position
  const float inv = __powf(10000.0f, -(float)i * (1.0f / 64.0f));
  float sn, cs;
  __sincosf((float)s * inv, &sn, &cs);
  __hip_bfloat16* p = q + (size_t)idx * 2;
  const float x0 = __bfloat162float(p[0]);
  const float x1 = __bfloat162float(p[1]);
  p[0] = __float2bfloat16(x0 * cs - x1 * sn);
  p[1] = __float2bfloat16(x0 * sn + x1 * cs);
}

// ---------------- Flash attention (causal GQA), WMMA bf16 ----------------
// Block: one (b, head) x 128 query rows; 8 waves x 16 query rows each.
// Key blocks of 32, double-buffered; K staged via async DMA, V^T via scatter.
__global__ __launch_bounds__(256, 1)
void attn_kernel(const __hip_bfloat16* __restrict__ q, const __hip_bfloat16* __restrict__ k,
                 const __hip_bfloat16* __restrict__ v, __hip_bfloat16* __restrict__ out) {
  __shared__ __hip_bfloat16 sK[2][32][HDIM + 8];  // [key][hd]   (272B rows)
  __shared__ __hip_bfloat16 sVt[2][HDIM][LDT];    // [hd][key]
  __shared__ __hip_bfloat16 sP[8][16][LDT];       // per-wave P tile [q][key]

  const int t = threadIdx.x;
  const int wv = t >> 5, lane = t & 31, half = lane >> 4, ln = lane & 15;
  const int b = blockIdx.z, h = blockIdx.y, g = h >> 2;   // h / QPK
  const int qb0 = blockIdx.x * 128;
  const int qrow = qb0 + wv * 16;

  // Q fragments: row s = qrow + ln, 4 chunks of HD (K=32 each)
  Frag16 qf[4];
  {
    const __hip_bfloat16* qp = q + (((size_t)b * SS + (qrow + ln)) * HQN + h) * HDIM;
#pragma unroll
    for (int c = 0; c < 4; c++) {
      qf[c].u[0] = *(const u32x4*)(qp + c * 32 + half * 8);
      qf[c].u[1] = *(const u32x4*)(qp + c * 32 + 16 + half * 8);
    }
  }

  f32x8 acc[8] = {};
  float mi[8], li[8];
#pragma unroll
  for (int r = 0; r < 8; r++) { mi[r] = -1.0e30f; li[r] = 0.f; }
  const float scale = 0.08838834764831845f;   // 1/sqrt(128)

  auto stage = [&](int t0, int nb) {
    const int kr = t >> 3, c0 = (t & 7) * 16;
    // K tile 32x128 row-major: async DMA (32B per thread).
    {
      const __hip_bfloat16* sbase = k + (((size_t)b * SS + t0) * HKVN + g) * HDIM;
      const unsigned voff = (unsigned)((kr * (HKVN * HDIM) + c0) * 2);
      const unsigned ldsb = (unsigned)(size_t)&sK[nb][kr][c0];
      asm volatile(
        "global_load_async_to_lds_b128 %0, %1, %2 offset:0\n\t"
        "global_load_async_to_lds_b128 %0, %1, %2 offset:16"
        :: "v"(ldsb), "v"(voff), "s"(sbase) : "memory");
    }
    // V tile transposed -> [hd][key]: manual scatter.
    {
      const size_t base = (((size_t)b * SS + (t0 + kr)) * HKVN + g) * HDIM + c0;
      u32x4 w0 = *(const u32x4*)(v + base);
      u32x4 w1 = *(const u32x4*)(v + base + 8);
      alignas(16) __hip_bfloat16 tmp[16];
      *(u32x4*)&tmp[0] = w0; *(u32x4*)&tmp[8] = w1;
#pragma unroll
      for (int i2 = 0; i2 < 16; i2++) sVt[nb][c0 + i2][kr] = tmp[i2];
    }
  };

  stage(0, 0);
  for (int t0 = 0; t0 < qb0 + 128; t0 += 32) {
    const int cur = (t0 >> 5) & 1;
    wait_async0();
    __syncthreads();
    if (t0 + 32 < qb0 + 128) stage(t0 + 32, cur ^ 1);

    if (t0 <= qrow + 15) {   // wave-uniform: block intersects causal region
      // scores: two 16x16 key tiles, contract over HD (4 WMMA K-steps each)
      f32x8 s0 = {}, s1 = {};
#pragma unroll
      for (int c = 0; c < 4; c++) {
        Frag16 b0, b1;
        const __hip_bfloat16* p0 = &sK[cur][ln][c * 32 + half * 8];
        b0.u[0] = *(const u32x4*)p0; b0.u[1] = *(const u32x4*)(p0 + 16);
        const __hip_bfloat16* p1 = &sK[cur][ln + 16][c * 32 + half * 8];
        b1.u[0] = *(const u32x4*)p1; b1.u[1] = *(const u32x4*)(p1 + 16);
        s0 = wmma_bf16(qf[c].v, b0.v, s0);
        s1 = wmma_bf16(qf[c].v, b1.v, s1);
      }
      // online softmax, rows m = r + 8*half, key cols n=ln (tile0) / ln+16 (tile1)
      float e0v[8], e1v[8], al[8];
#pragma unroll
      for (int r = 0; r < 8; r++) {
        const int qg = qrow + r + 8 * half;
        float a0 = s0[r] * scale, a1 = s1[r] * scale;
        if (t0 + ln > qg)      a0 = -1.0e30f;
        if (t0 + 16 + ln > qg) a1 = -1.0e30f;
        float mx = fmaxf(a0, a1);
        mx = fmaxf(mx, __shfl_xor(mx, 1));
        mx = fmaxf(mx, __shfl_xor(mx, 2));
        mx = fmaxf(mx, __shfl_xor(mx, 4));
        mx = fmaxf(mx, __shfl_xor(mx, 8));
        const float mn = fmaxf(mi[r], mx);
        const float e0 = __expf(a0 - mn), e1 = __expf(a1 - mn);
        float sm = e0 + e1;
        sm += __shfl_xor(sm, 1);
        sm += __shfl_xor(sm, 2);
        sm += __shfl_xor(sm, 4);
        sm += __shfl_xor(sm, 8);
        al[r] = __expf(mi[r] - mn);
        li[r] = li[r] * al[r] + sm;
        mi[r] = mn;
        e0v[r] = e0; e1v[r] = e1;
      }
#pragma unroll
      for (int nc = 0; nc < 8; nc++)
#pragma unroll
        for (int r = 0; r < 8; r++) acc[nc][r] *= al[r];

      // P: C-layout -> LDS -> A-layout fragment
#pragma unroll
      for (int r = 0; r < 8; r++) {
        const int m = r + 8 * half;
        sP[wv][m][ln]      = __float2bfloat16(e0v[r]);
        sP[wv][m][ln + 16] = __float2bfloat16(e1v[r]);
      }
      asm volatile("s_wait_dscnt 0x0" ::: "memory");   // same-wave DS RAW
      Frag16 pf;
      const __hip_bfloat16* pp = &sP[wv][ln][half * 8];
      pf.u[0] = *(const u32x4*)pp;
      pf.u[1] = *(const u32x4*)(pp + 16);

      // PV: out[16 x HD] += P[16x32] @ V[32xHD]
#pragma unroll
      for (int nc = 0; nc < 8; nc++) {
        Frag16 bv;
        const __hip_bfloat16* vp = &sVt[cur][nc * 16 + ln][half * 8];
        bv.u[0] = *(const u32x4*)vp; bv.u[1] = *(const u32x4*)(vp + 16);
        acc[nc] = wmma_bf16(pf.v, bv.v, acc[nc]);
      }
    }
  }

  // epilogue: out[b, s, h, hd] = acc / l
#pragma unroll
  for (int nc = 0; nc < 8; nc++)
#pragma unroll
    for (int r = 0; r < 8; r++) {
      const int s_ = qrow + r + 8 * half;
      const float o = acc[nc][r] / li[r];
      out[(((size_t)b * SS + s_) * HQN + h) * HDIM + nc * 16 + ln] = __float2bfloat16(o);
    }
}

// ---------------- launch ----------------
extern "C" void kernel_launch(void* const* d_in, const int* in_sizes, int n_in,
                              void* d_out, int out_size, void* d_ws, size_t ws_size,
                              hipStream_t stream) {
  const __hip_bfloat16* x  = (const __hip_bfloat16*)d_in[0];
  const float*          gw = (const float*)d_in[1];
  const __hip_bfloat16* wq = (const __hip_bfloat16*)d_in[2];
  const __hip_bfloat16* wk = (const __hip_bfloat16*)d_in[3];
  const __hip_bfloat16* wv = (const __hip_bfloat16*)d_in[4];
  const __hip_bfloat16* wo = (const __hip_bfloat16*)d_in[5];
  __hip_bfloat16* out = (__hip_bfloat16*)d_out;

  const size_t E_full = (size_t)MROWS * DD;              // 8M elems
  const size_t E_kv   = (size_t)MROWS * (HKVN * HDIM);   // 2M elems
  __hip_bfloat16* xn = (__hip_bfloat16*)d_ws;
  __hip_bfloat16* qb = xn + E_full;
  __hip_bfloat16* kb = qb + E_full;
  __hip_bfloat16* vb = kb + E_kv;
  __hip_bfloat16* ao = vb + E_kv;

  // 1) LayerNorm
  ln_kernel<<<MROWS, 256, 0, stream>>>(x, gw, xn);

  // 2) QKV projections (WMMA bf16, async-to-LDS pipelined)
  gemm_bf16<<<dim3(MROWS / 256, DD / 128), 256, 0, stream>>>(xn, wq, qb, MROWS, DD, DD);
  gemm_bf16<<<dim3(MROWS / 256, (HKVN * HDIM) / 128), 256, 0, stream>>>(xn, wk, kb, MROWS, HKVN * HDIM, DD);
  gemm_bf16<<<dim3(MROWS / 256, (HKVN * HDIM) / 128), 256, 0, stream>>>(xn, wv, vb, MROWS, HKVN * HDIM, DD);

  // 3) RoPE on q and k (in-place, fp32)
  const int qpairs = BB * SS * HQN * (HDIM / 2);
  const int kpairs = BB * SS * HKVN * (HDIM / 2);
  rope_kernel<<<(qpairs + 255) / 256, 256, 0, stream>>>(qb, HQN, qpairs);
  rope_kernel<<<(kpairs + 255) / 256, 256, 0, stream>>>(kb, HKVN, kpairs);

  // 4) causal GQA flash attention
  attn_kernel<<<dim3(SS / 128, HQN, BB), 256, 0, stream>>>(qb, kb, vb, ao);

  // 5) output projection
  gemm_bf16<<<dim3(MROWS / 256, DD / 128), 256, 0, stream>>>(ao, wo, out, MROWS, DD, DD);
}
